// InvariantTransformer_88656714925190
// MI455X (gfx1250) — matrix operand order repaired and checked by reading, not compiled, not run on hardware
//
#include <hip/hip_runtime.h>
#include <hip/hip_bf16.h>

typedef __attribute__((ext_vector_type(16))) __bf16          v16bf;
typedef __attribute__((ext_vector_type(16))) unsigned short  v16u16;
typedef __attribute__((ext_vector_type(8)))  float           v8f;

#define WPB 8            // waves per block (256 threads, wave32)
#define LAYERS 5

// ---------- helpers ----------
__device__ __forceinline__ unsigned short f2bf(float f) {
  unsigned u = __builtin_bit_cast(unsigned, f);
  unsigned r = u + 0x7FFFu + ((u >> 16) & 1u);   // round-to-nearest-even
  return (unsigned short)(r >> 16);
}
__device__ __forceinline__ float silu_f(float x) { return x / (1.0f + __expf(-x)); }

__device__ __forceinline__ v8f wmma_bf16(v16u16 a, v16u16 b, v8f c) {
  return __builtin_amdgcn_wmma_f32_16x16x32_bf16(
      false, __builtin_bit_cast(v16bf, a),
      false, __builtin_bit_cast(v16bf, b),
      (short)0, c, false, false);
}

// A fragment (16x32 bf16, M on lanes&15, K-half on lane>>4) from a row of a
// feature tile in LDS. rowPtr -> stage[m][0], row length >= 32*(kc+1) halfs.
__device__ __forceinline__ v16u16 load_a_frag(const unsigned short* rowPtr, int kh, int kc) {
  v16u16 a;
  const unsigned short* p = rowPtr + kc * 32 + kh * 8;
#pragma unroll
  for (int i = 0; i < 4; ++i) {
    unsigned d0 = *(const unsigned*)(p + 2 * i);          // K = 2i..2i+1 (+8kh+32kc)
    a[2 * i]     = (unsigned short)(d0 & 0xffffu);
    a[2 * i + 1] = (unsigned short)(d0 >> 16);
    unsigned d1 = *(const unsigned*)(p + 16 + 2 * i);     // K = 16+2i..17+2i
    a[8 + 2 * i] = (unsigned short)(d1 & 0xffffu);
    a[9 + 2 * i] = (unsigned short)(d1 >> 16);
  }
  return a;
}

// B fragment (32x16 bf16, N on lanes&15, K-half on lane>>4: K = 8h+s / 16+8h+(s-8))
// from a [K][32] row-major bf16 weight array; N-tile nt selects cols 16nt..16nt+15.
__device__ __forceinline__ v16u16 load_b_frag(const unsigned short* W, int lane, int kc, int nt) {
  const int n = nt * 16 + (lane & 15), kh = lane >> 4;
  v16u16 b;
#pragma unroll
  for (int s = 0; s < 16; ++s) {
    int k = (s < 8) ? (8 * kh + s) : (16 + 8 * kh + (s - 8));
    b[s] = W[(32 * kc + k) * 32 + n];
  }
  return b;
}

// ---------- weight prep: fp32 -> bf16 (We1 padded 80->96 rows) ----------
__global__ void prep_weights_kernel(const float* __restrict__ We1, const float* __restrict__ We2,
                                    const float* __restrict__ Wu1, const float* __restrict__ Wu2,
                                    unsigned short* __restrict__ We1bf, unsigned short* __restrict__ We2bf,
                                    unsigned short* __restrict__ Wu1bf, unsigned short* __restrict__ Wu2bf) {
  int tid = blockIdx.x * blockDim.x + threadIdx.x;
  int stride = gridDim.x * blockDim.x;
  for (int i = tid; i < LAYERS * 96 * 32; i += stride) {
    int l = i / (96 * 32), r = (i / 32) % 96, c = i % 32;
    We1bf[i] = (r < 80) ? f2bf(We1[(l * 80 + r) * 32 + c]) : (unsigned short)0;
  }
  for (int i = tid; i < LAYERS * 32 * 32; i += stride) We2bf[i] = f2bf(We2[i]);
  for (int i = tid; i < LAYERS * 64 * 32; i += stride) Wu1bf[i] = f2bf(Wu1[i]);
  for (int i = tid; i < LAYERS * 32 * 32; i += stride) Wu2bf[i] = f2bf(Wu2[i]);
}

// ---------- RBF + degree (pos mean cancels in pos[row]-pos[col]) ----------
__global__ void rbf_deg_kernel(const float* __restrict__ pos, const int* __restrict__ ei,
                               unsigned short* __restrict__ rbf, float* __restrict__ deg, int E_) {
  int e = blockIdx.x * blockDim.x + threadIdx.x;
  if (e >= E_) return;
  int r = ei[e], c = ei[E_ + e];
  float dx = pos[2 * r] - pos[2 * c];
  float dy = pos[2 * r + 1] - pos[2 * c + 1];
  float d = sqrtf(dx * dx + dy * dy);
  float inv = 1.41421356237f / (d + 1e-8f);
  unsigned w[8];
#pragma unroll
  for (int i = 0; i < 8; ++i) {
    float f0 = (2 * i + 1) * 3.14159265358979f;
    float f1 = (2 * i + 2) * 3.14159265358979f;
    unsigned short lo = f2bf(__sinf(d * f0) * inv);
    unsigned short hi = f2bf(__sinf(d * f1) * inv);
    w[i] = (unsigned)lo | ((unsigned)hi << 16);
  }
  uint4* dst = (uint4*)(rbf + (size_t)e * 16);
  dst[0] = make_uint4(w[0], w[1], w[2], w[3]);
  dst[1] = make_uint4(w[4], w[5], w[6], w[7]);
  atomicAdd(&deg[r], 1.0f);
}

__global__ void inv_deg_kernel(const float* __restrict__ deg, float* __restrict__ invd, int N_) {
  int n = blockIdx.x * blockDim.x + threadIdx.x;
  if (n < N_) invd[n] = 1.0f / fmaxf(deg[n], 1.0f);
}

// ---------- node embedding: invariant feats(10) -> 64 -> 32 ----------
__global__ __launch_bounds__(256) void node_embed_kernel(
    const float* __restrict__ u, const float* __restrict__ v,
    const float* __restrict__ bnorm, const float* __restrict__ yforce,
    const float* __restrict__ Wn1, const float* __restrict__ bn1,
    const float* __restrict__ Wn2, const float* __restrict__ bn2,
    float* __restrict__ x, unsigned short* __restrict__ xbf, int N_) {
  __shared__ float sW1[10 * 64], sb1[64], sW2[64 * 32], sb2[32];
  for (int i = threadIdx.x; i < 640; i += 256)  sW1[i] = Wn1[i];
  for (int i = threadIdx.x; i < 64; i += 256)   sb1[i] = bn1[i];
  for (int i = threadIdx.x; i < 2048; i += 256) sW2[i] = Wn2[i];
  for (int i = threadIdx.x; i < 32; i += 256)   sb2[i] = bn2[i];
  __syncthreads();
  int n = blockIdx.x * 256 + threadIdx.x;
  if (n >= N_) return;
  float f[10];
#pragma unroll
  for (int t = 0; t < 4; ++t) f[t] = u[n * 4 + t];
#pragma unroll
  for (int t = 0; t < 4; ++t) {
    float a = v[n * 8 + 2 * t], b = v[n * 8 + 2 * t + 1];
    f[4 + t] = sqrtf(a * a + b * b);
  }
  { float a = bnorm[2 * n], b = bnorm[2 * n + 1];  f[8] = sqrtf(a * a + b * b); }
  { float a = yforce[2 * n], b = yforce[2 * n + 1]; f[9] = sqrtf(a * a + b * b); }
  float hb[64];
  for (int j = 0; j < 64; ++j) {
    float acc = sb1[j];
#pragma unroll
    for (int k = 0; k < 10; ++k) acc += f[k] * sW1[k * 64 + j];
    hb[j] = silu_f(acc);
  }
  for (int s = 0; s < 32; ++s) {
    float acc = sb2[s];
    for (int j = 0; j < 64; ++j) acc += hb[j] * sW2[j * 32 + s];
    x[(size_t)n * 32 + s] = acc;
    xbf[(size_t)n * 32 + s] = f2bf(acc);
  }
}

// ---------- edge messages: [x_row|x_col|rbf](80->96) -> 32 (silu) -> 32, scatter-add ----------
__global__ __launch_bounds__(256) void edge_msg_kernel(
    const unsigned short* __restrict__ xbf, const unsigned short* __restrict__ rbf,
    const int* __restrict__ rowI, const int* __restrict__ colI,
    const unsigned short* __restrict__ We1bf, const unsigned short* __restrict__ We2bf,
    const float* __restrict__ be1, const float* __restrict__ be2,
    float* __restrict__ agg, int E_) {
  __shared__ unsigned short stage[WPB][16 * 96];
  __shared__ unsigned short hstage[WPB][16 * 32];
  __shared__ int rows_s[WPB][16];
  const int lane = threadIdx.x & 31;
  const int wave = threadIdx.x >> 5;
  const int m = lane & 15, h = lane >> 4;

  // preload weight B-fragments into registers (fixed for whole kernel)
  v16u16 bw1[3][2], bw2[2];
#pragma unroll
  for (int kc = 0; kc < 3; ++kc)
#pragma unroll
    for (int nt = 0; nt < 2; ++nt) bw1[kc][nt] = load_b_frag(We1bf, lane, kc, nt);
#pragma unroll
  for (int nt = 0; nt < 2; ++nt) bw2[nt] = load_b_frag(We2bf, lane, 0, nt);
  const float be1v0 = be1[m], be1v1 = be1[m + 16];
  const float be2v0 = be2[m], be2v1 = be2[m + 16];

  if (h == 0) {  // zero K-padding region 80..95 once (never overwritten)
#pragma unroll
    for (int j = 0; j < 16; ++j) stage[wave][m * 96 + 80 + j] = 0;
  }

  const int numTiles = (E_ + 15) >> 4;
  for (int tile = blockIdx.x * WPB + wave; tile < numTiles; tile += gridDim.x * WPB) {
    const int base = tile << 4;
    int e = base + m; if (e >= E_) e = E_ - 1;
    const int rn = rowI[e], cn = colI[e];
    if (h == 0) rows_s[wave][m] = rn;
    {  // gather features -> LDS tile (per-wave private; same-wave DS is in order)
      const uint4* xr = (const uint4*)(xbf + (size_t)rn * 32 + h * 16);
      uint4* d0 = (uint4*)&stage[wave][m * 96 + h * 16];
      d0[0] = xr[0]; d0[1] = xr[1];
      const uint4* xc = (const uint4*)(xbf + (size_t)cn * 32 + h * 16);
      uint4* d1 = (uint4*)&stage[wave][m * 96 + 32 + h * 16];
      d1[0] = xc[0]; d1[1] = xc[1];
      const uint4* rb = (const uint4*)(rbf + (size_t)e * 16 + h * 8);
      *(uint4*)&stage[wave][m * 96 + 64 + h * 8] = rb[0];
    }
    // GEMM1: [16x96] x [96x32]
    v8f acc0 = {}, acc1 = {};
    const unsigned short* rowPtr = &stage[wave][m * 96];
#pragma unroll
    for (int kc = 0; kc < 3; ++kc) {
      v16u16 a = load_a_frag(rowPtr, h, kc);
      acc0 = wmma_bf16(a, bw1[kc][0], acc0);
      acc1 = wmma_bf16(a, bw1[kc][1], acc1);
    }
    // bias + SiLU, restage D-layout -> feature tile for GEMM2
#pragma unroll
    for (int r = 0; r < 8; ++r) {
      int mm = r + 8 * h;  // D layout: M = r + 8*(lane>>4), N = lane&15 (+16*nt)
      hstage[wave][mm * 32 + m]      = f2bf(silu_f(acc0[r] + be1v0));
      hstage[wave][mm * 32 + m + 16] = f2bf(silu_f(acc1[r] + be1v1));
    }
    // GEMM2: [16x32] x [32x32]
    v8f o0 = {}, o1 = {};
    {
      v16u16 a2 = load_a_frag(&hstage[wave][m * 32], h, 0);
      o0 = wmma_bf16(a2, bw2[0], o0);
      o1 = wmma_bf16(a2, bw2[1], o1);
    }
    // scatter-add messages into agg[row]
#pragma unroll
    for (int r = 0; r < 8; ++r) {
      int mm = r + 8 * h;
      int e2 = base + mm;
      if (e2 < E_) {
        int node = rows_s[wave][mm];
        atomicAdd(&agg[(size_t)node * 32 + m],      o0[r] + be2v0);
        atomicAdd(&agg[(size_t)node * 32 + m + 16], o1[r] + be2v1);
      }
    }
  }
}

// ---------- node update: [x|agg*inv_deg](64) -> 32 (silu) -> 32, residual ----------
__global__ __launch_bounds__(256) void node_update_kernel(
    float* __restrict__ x, unsigned short* __restrict__ xbf,
    const float* __restrict__ agg, const float* __restrict__ invd,
    const unsigned short* __restrict__ Wu1bf, const unsigned short* __restrict__ Wu2bf,
    const float* __restrict__ bu1, const float* __restrict__ bu2, int N_) {
  __shared__ unsigned short stage[WPB][16 * 64];
  __shared__ unsigned short hstage[WPB][16 * 32];
  const int lane = threadIdx.x & 31;
  const int wave = threadIdx.x >> 5;
  const int m = lane & 15, h = lane >> 4;

  v16u16 bw1[2][2], bw2[2];
#pragma unroll
  for (int kc = 0; kc < 2; ++kc)
#pragma unroll
    for (int nt = 0; nt < 2; ++nt) bw1[kc][nt] = load_b_frag(Wu1bf, lane, kc, nt);
#pragma unroll
  for (int nt = 0; nt < 2; ++nt) bw2[nt] = load_b_frag(Wu2bf, lane, 0, nt);
  const float bu1v0 = bu1[m], bu1v1 = bu1[m + 16];
  const float bu2v0 = bu2[m], bu2v1 = bu2[m + 16];

  const int numTiles = (N_ + 15) >> 4;
  for (int tile = blockIdx.x * WPB + wave; tile < numTiles; tile += gridDim.x * WPB) {
    const int base = tile << 4;
    int n = base + m; if (n >= N_) n = N_ - 1;
    {  // features 0..31 = x (bf16), 32..63 = agg * inv_deg
      const uint4* xr = (const uint4*)(xbf + (size_t)n * 32 + h * 16);
      uint4* d0 = (uint4*)&stage[wave][m * 64 + h * 16];
      d0[0] = xr[0]; d0[1] = xr[1];
      const float s = invd[n];
      const float* ag = agg + (size_t)n * 32 + h * 16;
      unsigned w[8];
#pragma unroll
      for (int i = 0; i < 8; ++i) {
        unsigned short lo = f2bf(ag[2 * i] * s);
        unsigned short hi = f2bf(ag[2 * i + 1] * s);
        w[i] = (unsigned)lo | ((unsigned)hi << 16);
      }
      uint4* d1 = (uint4*)&stage[wave][m * 64 + 32 + h * 16];
      d1[0] = make_uint4(w[0], w[1], w[2], w[3]);
      d1[1] = make_uint4(w[4], w[5], w[6], w[7]);
    }
    v8f acc0 = {}, acc1 = {};
    const unsigned short* rowPtr = &stage[wave][m * 64];
#pragma unroll
    for (int kc = 0; kc < 2; ++kc) {
      v16u16 a = load_a_frag(rowPtr, h, kc);
      acc0 = wmma_bf16(a, bw1[kc][0], acc0);
      acc1 = wmma_bf16(a, bw1[kc][1], acc1);
    }
#pragma unroll
    for (int r = 0; r < 8; ++r) {
      int mm = r + 8 * h;
      hstage[wave][mm * 32 + m]      = f2bf(silu_f(acc0[r] + bu1v0));
      hstage[wave][mm * 32 + m + 16] = f2bf(silu_f(acc1[r] + bu1v1));
    }
    v8f o0 = {}, o1 = {};
    {
      v16u16 a2 = load_a_frag(&hstage[wave][m * 32], h, 0);
      o0 = wmma_bf16(a2, bw2[0], o0);
      o1 = wmma_bf16(a2, bw2[1], o1);
    }
#pragma unroll
    for (int r = 0; r < 8; ++r) {
      int mm = r + 8 * h;
      int n2 = base + mm;
      if (n2 < N_) {
        size_t i0 = (size_t)n2 * 32 + m;
        float nv0 = x[i0] + o0[r] + bu2v0;
        float nv1 = x[i0 + 16] + o1[r] + bu2v1;
        x[i0] = nv0;        x[i0 + 16] = nv1;
        xbf[i0] = f2bf(nv0); xbf[i0 + 16] = f2bf(nv1);
      }
    }
  }
}

// ---------- LayerNorm + heads ----------
__global__ __launch_bounds__(256) void head_kernel(
    const float* __restrict__ x, const float* __restrict__ u, const float* __restrict__ v,
    const float* __restrict__ ln_g, const float* __restrict__ ln_b,
    const float* __restrict__ Wr1, const float* __restrict__ br1,
    const float* __restrict__ Wr2, const float* __restrict__ br2,
    const float* __restrict__ Ws1, const float* __restrict__ bs1,
    const float* __restrict__ Ws2, const float* __restrict__ bs2,
    float* __restrict__ out, int N_) {
  __shared__ float sg[32], sb[32], sWr1[32 * 96], sbr1[96], sWr2[96 * 2], sbr2[2];
  __shared__ float sWs1[32 * 96], sbs1[96], sWs2[96], sbs2[1];
  for (int i = threadIdx.x; i < 32; i += 256)   { sg[i] = ln_g[i]; sb[i] = ln_b[i]; }
  for (int i = threadIdx.x; i < 3072; i += 256) { sWr1[i] = Wr1[i]; sWs1[i] = Ws1[i]; }
  for (int i = threadIdx.x; i < 96; i += 256)   { sbr1[i] = br1[i]; sbs1[i] = bs1[i]; sWs2[i] = Ws2[i]; }
  for (int i = threadIdx.x; i < 192; i += 256)  sWr2[i] = Wr2[i];
  if (threadIdx.x < 2) sbr2[threadIdx.x] = br2[threadIdx.x];
  if (threadIdx.x == 0) sbs2[0] = bs2[0];
  __syncthreads();
  int n = blockIdx.x * 256 + threadIdx.x;
  if (n >= N_) return;
  float xv[32];
  float mu = 0.f;
#pragma unroll
  for (int s = 0; s < 32; ++s) { xv[s] = x[(size_t)n * 32 + s]; mu += xv[s]; }
  mu *= (1.0f / 32.0f);
  float var = 0.f;
#pragma unroll
  for (int s = 0; s < 32; ++s) { float d = xv[s] - mu; var += d * d; }
  var *= (1.0f / 32.0f);
  float rs = rsqrtf(var + 1e-5f);
  float xn[32];
#pragma unroll
  for (int s = 0; s < 32; ++s) xn[s] = (xv[s] - mu) * rs * sg[s] + sb[s];
  float a0 = sbr2[0], a1 = sbr2[1], as_ = sbs2[0];
  for (int j = 0; j < 96; ++j) {
    float hr = sbr1[j], hs = sbs1[j];
#pragma unroll
    for (int s = 0; s < 32; ++s) {
      hr += xn[s] * sWr1[s * 96 + j];
      hs += xn[s] * sWs1[s * 96 + j];
    }
    hr = silu_f(hr); hs = silu_f(hs);
    a0 += hr * sWr2[j * 2];
    a1 += hr * sWr2[j * 2 + 1];
    as_ += hs * sWs2[j];
  }
  out[n] = u[n * 4 + 3] + as_;                 // u[:, -1] + u_[:, 0]
  out[N_ + 2 * n]     = a0 + v[n * 8 + 6];     // v_ + v[:, -1, :]
  out[N_ + 2 * n + 1] = a1 + v[n * 8 + 7];
}

// ---------- host launcher ----------
extern "C" void kernel_launch(void* const* d_in, const int* in_sizes, int n_in,
                              void* d_out, int out_size, void* d_ws, size_t ws_size,
                              hipStream_t stream) {
  const float* u       = (const float*)d_in[0];
  const float* v       = (const float*)d_in[1];
  const float* bnorm   = (const float*)d_in[2];
  const float* yforce  = (const float*)d_in[4];
  const float* pos     = (const float*)d_in[5];
  const int*   ei      = (const int*)d_in[6];
  const float* Wn1 = (const float*)d_in[7],  *bn1 = (const float*)d_in[8];
  const float* Wn2 = (const float*)d_in[9],  *bn2 = (const float*)d_in[10];
  const float* We1 = (const float*)d_in[11], *be1 = (const float*)d_in[12];
  const float* We2 = (const float*)d_in[13], *be2 = (const float*)d_in[14];
  const float* Wu1 = (const float*)d_in[15], *bu1 = (const float*)d_in[16];
  const float* Wu2 = (const float*)d_in[17], *bu2 = (const float*)d_in[18];
  const float* ln_g = (const float*)d_in[19], *ln_b = (const float*)d_in[20];
  const float* Wr1 = (const float*)d_in[21], *br1 = (const float*)d_in[22];
  const float* Wr2 = (const float*)d_in[23], *br2 = (const float*)d_in[24];
  const float* Ws1 = (const float*)d_in[25], *bs1 = (const float*)d_in[26];
  const float* Ws2 = (const float*)d_in[27], *bs2 = (const float*)d_in[28];
  float* out = (float*)d_out;

  const int N = in_sizes[0] / 4;   // u is [N,4]
  const int E = in_sizes[6] / 2;   // edge_index is [2,E]

  char* ws = (char*)d_ws;
  size_t off = 0;
  auto carve = [&](size_t bytes) -> void* {
    void* p = ws + off;
    off = (off + bytes + 255) & ~(size_t)255;
    return p;
  };
  float*          x     = (float*)carve((size_t)N * 32 * 4);
  unsigned short* xbf   = (unsigned short*)carve((size_t)N * 32 * 2);
  float*          agg   = (float*)carve((size_t)N * 32 * 4);
  float*          deg   = (float*)carve((size_t)N * 4);
  float*          invd  = (float*)carve((size_t)N * 4);
  unsigned short* rbf   = (unsigned short*)carve((size_t)E * 16 * 2);
  unsigned short* We1bf = (unsigned short*)carve((size_t)LAYERS * 96 * 32 * 2);
  unsigned short* We2bf = (unsigned short*)carve((size_t)LAYERS * 32 * 32 * 2);
  unsigned short* Wu1bf = (unsigned short*)carve((size_t)LAYERS * 64 * 32 * 2);
  unsigned short* Wu2bf = (unsigned short*)carve((size_t)LAYERS * 32 * 32 * 2);

  prep_weights_kernel<<<64, 256, 0, stream>>>(We1, We2, Wu1, Wu2, We1bf, We2bf, Wu1bf, Wu2bf);

  hipMemsetAsync(deg, 0, (size_t)N * 4, stream);
  rbf_deg_kernel<<<(E + 255) / 256, 256, 0, stream>>>(pos, ei, rbf, deg, E);
  inv_deg_kernel<<<(N + 255) / 256, 256, 0, stream>>>(deg, invd, N);

  node_embed_kernel<<<(N + 255) / 256, 256, 0, stream>>>(u, v, bnorm, yforce,
                                                         Wn1, bn1, Wn2, bn2, x, xbf, N);

  const int numTilesE = (E + 15) / 16;
  const int numTilesN = (N + 15) / 16;
  int gridE = (numTilesE + WPB - 1) / WPB; if (gridE > 2048) gridE = 2048;
  int gridN = (numTilesN + WPB - 1) / WPB; if (gridN > 2048) gridN = 2048;

  for (int l = 0; l < LAYERS; ++l) {
    hipMemsetAsync(agg, 0, (size_t)N * 32 * 4, stream);
    edge_msg_kernel<<<gridE, 256, 0, stream>>>(
        xbf, rbf, ei, ei + E,
        We1bf + (size_t)l * 96 * 32, We2bf + (size_t)l * 32 * 32,
        be1 + l * 32, be2 + l * 32, agg, E);
    node_update_kernel<<<gridN, 256, 0, stream>>>(
        x, xbf, agg, invd,
        Wu1bf + (size_t)l * 64 * 32, Wu2bf + (size_t)l * 32 * 32,
        bu1 + l * 32, bu2 + l * 32, N);
  }

  head_kernel<<<(N + 255) / 256, 256, 0, stream>>>(x, u, v, ln_g, ln_b,
                                                   Wr1, br1, Wr2, br2,
                                                   Ws1, bs1, Ws2, bs2, out, N);
}
